// GraphAttentionLayer_1108101562597
// MI455X (gfx1250) — matrix operand order repaired and checked by reading, not compiled, not run on hardware
//
#include <hip/hip_runtime.h>
#include <hip/hip_bf16.h>

typedef __attribute__((ext_vector_type(16))) _Float16 v16h;
typedef __attribute__((ext_vector_type(8)))  float    v8f;

#define ALPHA 0.2f
#define NEG_BIG (-9000000000000000.0f)
#define BS 4
#define NN 4096
#define FF 256
#define ROWS (BS * NN)          // 16384

// -------- workspace layout (bytes) --------
#define OFF_H32  ((size_t)0)                                   // 16384*256 f32 = 16 MB
#define OFF_H16  (OFF_H32 + (size_t)ROWS * FF * 4)             // 16384*256 f16 = 8 MB
#define OFF_F1   (OFF_H16 + (size_t)ROWS * FF * 2)             // 16384 f32
#define OFF_F2   (OFF_F1 + (size_t)ROWS * 4)
#define OFF_CMAX (OFF_F2 + (size_t)ROWS * 4)
#define OFF_RCS  (OFF_CMAX + (size_t)ROWS * 4)

// -------- gfx1250 async global->LDS support (guarded) --------
#if defined(__gfx1250__) && __has_builtin(__builtin_amdgcn_global_load_async_to_lds_b128)
#define HAS_ASYNC 1
#else
#define HAS_ASYNC 0
#endif

#if HAS_ASYNC
  #if __has_builtin(__builtin_amdgcn_s_wait_asynccnt)
    #define WAIT_ASYNC() __builtin_amdgcn_s_wait_asynccnt(0)
  #else
    #define WAIT_ASYNC() asm volatile("s_wait_asynccnt 0x0" ::: "memory")
  #endif
#else
  #define WAIT_ASYNC() do {} while (0)
#endif

// builtin prototype (from hipcc diagnostic): (v4i AS1*, v4i AS3*, int off, int cpol)
typedef int gld_v4i __attribute__((vector_size(16)));
#define AS1V(p) ((__attribute__((address_space(1))) gld_v4i*)(uintptr_t)(p))
#define AS3V(p) ((__attribute__((address_space(3))) gld_v4i*)(uintptr_t)(p))

// =====================================================================
// Kernel 1: h = X @ W   (16384x256 = 16384x256 * 256x256), f16 WMMA,
// f32 accumulate. Store h as f32 and f16. One wave per 16x16 tile.
// =====================================================================
__global__ __launch_bounds__(256)
void gat_gemm1(const float* __restrict__ X, const float* __restrict__ W,
               float* __restrict__ h32, _Float16* __restrict__ h16) {
  const int tid  = threadIdx.x;
  const int lane = tid & 31;
  const int half = lane >> 4;
  const int l16  = lane & 15;
  const int wid  = blockIdx.x * 8 + (tid >> 5);
  const int tf   = wid & 15;
  const int tr   = wid >> 4;
  const int r0   = tr * 16;
  const int f0   = tf * 16;

  const int arow = r0 + l16;
  const int fcol = f0 + l16;

  v8f c = {};
  for (int kc = 0; kc < FF; kc += 32) {
    v16h a;   // a[i] <-> K = 16*half + i
    const float* ap = X + (size_t)arow * FF + kc;
    #pragma unroll
    for (int g = 0; g < 2; ++g) {
      const float* base = ap + 16 * half + 8 * g;
      #pragma unroll
      for (int j = 0; j < 8; ++j) a[8 * g + j] = (_Float16)base[j];
    }
    v16h b;   // b[i] <-> K = 16*half + i
    #pragma unroll
    for (int i = 0; i < 16; ++i)
      b[i] = (_Float16)W[(size_t)(kc + 16 * half + i) * FF + fcol];

    c = __builtin_amdgcn_wmma_f32_16x16x32_f16(false, a, false, b,
                                               (short)0, c, false, false);
  }
  #pragma unroll
  for (int r = 0; r < 8; ++r) {
    const int rowo = r0 + 8 * half + r;
    const size_t idx = (size_t)rowo * FF + fcol;
    h32[idx] = c[r];
    h16[idx] = (_Float16)c[r];
  }
}

// =====================================================================
// Kernel 2: f1 = h·a1, f2 = h·a2. One wave per row, shuffle reduce.
// =====================================================================
__global__ __launch_bounds__(256)
void gat_proj(const float* __restrict__ h32, const float* __restrict__ avec,
              float* __restrict__ f1, float* __restrict__ f2) {
  const int tid  = threadIdx.x;
  const int lane = tid & 31;
  const int row  = blockIdx.x * 8 + (tid >> 5);
  const float* hp = h32 + (size_t)row * FF + lane * 8;
  const float* a1 = avec + lane * 8;
  const float* a2 = avec + 256 + lane * 8;
  float s1 = 0.f, s2 = 0.f;
  #pragma unroll
  for (int j = 0; j < 8; ++j) {
    const float hv = hp[j];
    s1 += hv * a1[j];
    s2 += hv * a2[j];
  }
  #pragma unroll
  for (int m = 16; m >= 1; m >>= 1) {
    s1 += __shfl_xor(s1, m, 32);
    s2 += __shfl_xor(s2, m, 32);
  }
  if (lane == 0) { f1[row] = s1; f2[row] = s2; }
}

// =====================================================================
// Kernel 3: per-(b,m) column online-softmax stats over n (axis=1).
// =====================================================================
__global__ __launch_bounds__(256)
void gat_colstats(const int* __restrict__ adj, const float* __restrict__ f1,
                  const float* __restrict__ f2, float* __restrict__ cmax,
                  float* __restrict__ rcs) {
  __shared__ float sM[256], sS[256];
  const int tid   = threadIdx.x;
  const int b     = blockIdx.x >> 6;
  const int m0    = (blockIdx.x & 63) * 64;
  const int mloc  = tid & 63;
  const int slice = tid >> 6;
  const int m     = m0 + mloc;

  const float f2v = f2[b * NN + m];
  const int* arow = adj + (size_t)b * NN * NN + m;
  float M = -__builtin_huge_valf();
  float S = 0.f;
  for (int n = slice; n < NN; n += 4) {
    __builtin_prefetch(arow + (size_t)(n + 32) * NN, 0, 0);   // stream ahead
    const int ad = arow[(size_t)n * NN];
    const float x = f1[b * NN + n] + f2v;
    const float e = x > 0.f ? x : ALPHA * x;
    const float l = ad > 0 ? e : NEG_BIG;
    if (l > M) { S = S * __expf(M - l) + 1.0f; M = l; }
    else       { S += __expf(l - M); }
  }
  sM[tid] = M; sS[tid] = S;
  __syncthreads();
  if (tid < 64) {
    float Mo = sM[tid], So = sS[tid];
    #pragma unroll
    for (int k = 1; k < 4; ++k) {
      const float Mi = sM[tid + 64 * k], Si = sS[tid + 64 * k];
      const float Mn = Mo > Mi ? Mo : Mi;
      So = So * __expf(Mo - Mn) + Si * __expf(Mi - Mn);
      Mo = Mn;
    }
    cmax[b * NN + m0 + tid] = Mo;
    rcs [b * NN + m0 + tid] = 1.0f / So;
  }
}

// =====================================================================
// Kernel 4: out = elu( softmax_col(adj-masked logits) @ h )
// Block tile 32n x 256f, m-chunks of 32. Double-buffered A (packed WMMA
// register layout in LDS) and B (32x256 f16 panel, async global->LDS).
// One barrier + one s_wait_asynccnt per chunk.
// =====================================================================
#define BP  264          // padded B row stride in halves (528B, 16B aligned)
#define NCH (NN / 32)    // 128 chunks

__device__ __forceinline__
void gen_A(int mc, _Float16* __restrict__ dst, const int* __restrict__ adjrow,
           float f1v, const float* __restrict__ f2, const float* __restrict__ cmax,
           const float* __restrict__ rcs, int bN, int s0, int gkbase) {
  #pragma unroll
  for (int e = 0; e < 4; ++e) {
    const int m  = mc + gkbase + e;          // 4 contiguous m per thread
    __builtin_prefetch(adjrow + m + 256, 0, 0);   // 8 chunks ahead
    const int ad = adjrow[m];
    const float x  = f1v + f2[bN + m];
    const float lr = x > 0.f ? x : ALPHA * x;
    float Av = 0.f;
    if (ad > 0) Av = __expf(lr - cmax[bN + m]) * rcs[bN + m];
    dst[s0 + e] = (_Float16)Av;
  }
}

__device__ __forceinline__
void stage_B(int mc, _Float16* __restrict__ dst, const _Float16* __restrict__ h16,
             int bN, int w, int lane) {
#if HAS_ASYNC
  // 8 waves x 4 rows: each async b128 moves one 512B row (32 lanes x 16B).
  #pragma unroll
  for (int r4 = 0; r4 < 4; ++r4) {
    const int row = w * 4 + r4;
    const _Float16* gp = h16 + ((size_t)bN + mc + row) * FF + lane * 8;
    _Float16* lp = dst + row * BP + lane * 8;
    __builtin_amdgcn_global_load_async_to_lds_b128(AS1V(gp), AS3V(lp), 0, 0);
  }
#else
  const int tid = w * 32 + lane;
  #pragma unroll
  for (int j = 0; j < 4; ++j) {
    const int vv  = tid * 4 + j;
    const int row = vv >> 5;
    const int c8  = vv & 31;
    const uint4 v = *(const uint4*)(h16 + ((size_t)bN + mc + row) * FF + c8 * 8);
    *(uint4*)(dst + row * BP + c8 * 8) = v;
  }
#endif
}

__global__ __launch_bounds__(256)
void gat_attn_gemm(const int* __restrict__ adj, const _Float16* __restrict__ h16,
                   const float* __restrict__ f1, const float* __restrict__ f2,
                   const float* __restrict__ cmax, const float* __restrict__ rcs,
                   float* __restrict__ out) {
  __shared__ __align__(32) _Float16 Apack[2][1024];
  __shared__ __align__(16) _Float16 Bt[2][32 * BP];

  const int tid  = threadIdx.x;
  const int lane = tid & 31;
  const int l16  = lane & 15;
  const int half = lane >> 4;
  const int w    = tid >> 5;
  const int sub  = w & 1;
  const int fbase = (w >> 1) * 64;

  const int b  = blockIdx.x >> 7;
  const int n0 = (blockIdx.x & 127) * 32;
  const int bN = b * NN;

  // A-generation constants: thread t owns packed halves [4t, 4t+4)
  const int s0     = tid * 4;
  const int gsub   = s0 >> 9;
  const int glane  = (s0 >> 4) & 31;
  const int gi0    = s0 & 15;
  const int gn     = n0 + gsub * 16 + (glane & 15);
  const int gkbase = 16 * (glane >> 4) + gi0;
  const float f1v  = f1[bN + gn];
  const int* adjrow = adj + ((size_t)bN + gn) * NN;

  v8f c[4] = {v8f{}, v8f{}, v8f{}, v8f{}};

  // prologue: chunk 0 into buffer 0
  gen_A(0, &Apack[0][0], adjrow, f1v, f2, cmax, rcs, bN, s0, gkbase);
  stage_B(0, &Bt[0][0], h16, bN, w, lane);

  for (int k = 0; k < NCH; ++k) {
    const int buf = k & 1;
    WAIT_ASYNC();        // my async B_k rows have landed
    __syncthreads();     // everyone's A_k stores + B_k rows visible

    if (k + 1 < NCH) {   // overlap: next B panel in flight + next A gen
      stage_B((k + 1) * 32, &Bt[buf ^ 1][0], h16, bN, w, lane);
      gen_A((k + 1) * 32, &Apack[buf ^ 1][0], adjrow, f1v, f2, cmax, rcs,
            bN, s0, gkbase);
    }

    const v16h a = *(const v16h*)&Apack[buf][(sub * 32 + lane) * 16];
    #pragma unroll
    for (int ft = 0; ft < 4; ++ft) {
      const int fc = fbase + ft * 16 + l16;
      v16h bv;   // b[i] <-> K = 16*half + i
      #pragma unroll
      for (int i = 0; i < 16; ++i)
        bv[i] = Bt[buf][(16 * half + i) * BP + fc];
      c[ft] = __builtin_amdgcn_wmma_f32_16x16x32_f16(false, a, false, bv,
                                                     (short)0, c[ft], false, false);
    }
  }

  // ELU epilogue
  #pragma unroll
  for (int ft = 0; ft < 4; ++ft) {
    const int fc = fbase + ft * 16 + l16;
    #pragma unroll
    for (int r = 0; r < 8; ++r) {
      const int n = n0 + sub * 16 + 8 * half + r;
      const float v = c[ft][r];
      out[((size_t)bN + n) * FF + fc] = v > 0.f ? v : (__expf(v) - 1.0f);
    }
  }
}

// =====================================================================
extern "C" void kernel_launch(void* const* d_in, const int* in_sizes, int n_in,
                              void* d_out, int out_size, void* d_ws, size_t ws_size,
                              hipStream_t stream) {
  const float* X    = (const float*)d_in[0];   // (4,4096,256)
  const int*   adj  = (const int*)d_in[1];     // (4,4096,4096)
  const float* W    = (const float*)d_in[2];   // (256,256)
  const float* avec = (const float*)d_in[3];   // (512,)
  float* out = (float*)d_out;

  char* ws = (char*)d_ws;
  float*    h32  = (float*)(ws + OFF_H32);
  _Float16* h16  = (_Float16*)(ws + OFF_H16);
  float*    f1   = (float*)(ws + OFF_F1);
  float*    f2   = (float*)(ws + OFF_F2);
  float*    cmax = (float*)(ws + OFF_CMAX);
  float*    rcs  = (float*)(ws + OFF_RCS);

  gat_gemm1<<<2048, 256, 0, stream>>>(X, W, h32, h16);
  gat_proj<<<2048, 256, 0, stream>>>(h32, avec, f1, f2);
  gat_colstats<<<BS * (NN / 64), 256, 0, stream>>>(adj, f1, f2, cmax, rcs);
  gat_attn_gemm<<<BS * (NN / 32), 256, 0, stream>>>(adj, h16, f1, f2, cmax, rcs, out);
}
